// LLoCaFramePredictor_84713934946690
// MI455X (gfx1250) — matrix operand order repaired and checked by reading, not compiled, not run on hardware
//
#include <hip/hip_runtime.h>
#include <hip/hip_bf16.h>

typedef __attribute__((ext_vector_type(16))) _Float16 v16h;
typedef __attribute__((ext_vector_type(8)))  _Float16 v8h;
typedef __attribute__((ext_vector_type(8)))  float    v8f;

#define EPS32 1.1920929e-07f
#define GAMMA_MAX 10.0f
#define MLP_T 8  // tiles (of 16 edges) processed per wave

// ---------------------------------------------------------------------------
// WMMA helpers (CDNA5 gfx1250, wave32)
// ---------------------------------------------------------------------------
__device__ __forceinline__ v8f wmma_f16(v16h a, v16h b, v8f c) {
  // D = A(16x32 f16) * B(32x16 f16) + C(16x16 f32)
  return __builtin_amdgcn_wmma_f32_16x16x32_f16(false, a, false, b, (short)0, c,
                                                false, false);
}

__device__ __forceinline__ v16h cat8(v8h lo, v8h hi) {
  return __builtin_shufflevector(lo, hi, 0, 1, 2, 3, 4, 5, 6, 7, 8, 9, 10, 11,
                                 12, 13, 14, 15);
}

// A-matrix 16x32 f16 layout (ISA 7.12.2): lane m=lane&15; lanes<16 hold
// K {0..7,16..23}, lanes>=16 hold K {8..15,24..31}  -> two contiguous 8-half
// chunks at (kbase+koff) and (kbase+koff+16).  Row stride in LDS = 64 halves.
__device__ __forceinline__ v16h load_a_frag(const _Float16* x, int kbase,
                                            int lane) {
  int m = lane & 15;
  int koff = (lane & 16) ? 8 : 0;
  const _Float16* p = x + m * 64 + kbase + koff;
  v8h lo = *(const v8h*)(p);
  v8h hi = *(const v8h*)(p + 16);
  return cat8(lo, hi);
}

// B fragments are pre-swizzled in LDS as [lane][16 halves] contiguous.
__device__ __forceinline__ v16h load_b_frag(const _Float16* ws, int lane) {
  const _Float16* p = ws + lane * 16;
  v8h lo = *(const v8h*)(p);
  v8h hi = *(const v8h*)(p + 8);
  return cat8(lo, hi);
}

// Zero-cost intra-wave fence: prevents the compiler from reordering LDS ops
// across cross-lane communication points; hardware LDS is in-order per wave.
__device__ __forceinline__ void wave_lds_fence() {
  __builtin_amdgcn_wave_barrier();
}

__device__ __forceinline__ float wave_sum(float v) {
#pragma unroll
  for (int o = 16; o > 0; o >>= 1) v += __shfl_xor(v, o, 32);
  return v;
}
__device__ __forceinline__ float wave_max(float v) {
#pragma unroll
  for (int o = 16; o > 0; o >>= 1) v = fmaxf(v, __shfl_xor(v, o, 32));
  return v;
}

// ---------------------------------------------------------------------------
// Kernel 1: per-edge Minkowski product + block partial sums (for mean/std)
// ---------------------------------------------------------------------------
__global__ void __launch_bounds__(256) llc_ea_kernel(
    const float* __restrict__ p4, const int* __restrict__ src,
    const int* __restrict__ dst, float* __restrict__ ea,
    double* __restrict__ partials, int E) {
  __shared__ double ssum[256];
  __shared__ double ssq[256];
  int tid = threadIdx.x;
  int e = blockIdx.x * 256 + tid;
  float v = 0.f;
  bool ok = e < E;
  if (ok) {
    int s = src[e], d = dst[e];
    float a0 = p4[s * 4 + 0], a1 = p4[s * 4 + 1], a2 = p4[s * 4 + 2],
          a3 = p4[s * 4 + 3];
    float c0 = p4[d * 4 + 0], c1 = p4[d * 4 + 1], c2 = p4[d * 4 + 2],
          c3 = p4[d * 4 + 3];
    v = a0 * c0 - a1 * c1 - a2 * c2 - a3 * c3;
    ea[e] = v;
  }
  double dv = ok ? (double)v : 0.0;
  ssum[tid] = dv;
  ssq[tid] = dv * dv;
  __syncthreads();
  for (int s2 = 128; s2 > 0; s2 >>= 1) {
    if (tid < s2) {
      ssum[tid] += ssum[tid + s2];
      ssq[tid] += ssq[tid + s2];
    }
    __syncthreads();
  }
  if (tid == 0) {
    partials[2 * blockIdx.x] = ssum[0];
    partials[2 * blockIdx.x + 1] = ssq[0];
  }
}

// ---------------------------------------------------------------------------
// Kernel 2: reduce partials -> mean, inv_std (ddof=1)
// ---------------------------------------------------------------------------
__global__ void __launch_bounds__(256) llc_stats_kernel(
    const double* __restrict__ partials, int nb, long long E,
    float* __restrict__ stats) {
  __shared__ double s0[256], s1[256];
  int tid = threadIdx.x;
  double a = 0.0, b = 0.0;
  for (int i = tid; i < nb; i += 256) {
    a += partials[2 * i];
    b += partials[2 * i + 1];
  }
  s0[tid] = a;
  s1[tid] = b;
  __syncthreads();
  for (int s = 128; s > 0; s >>= 1) {
    if (tid < s) {
      s0[tid] += s0[tid + s];
      s1[tid] += s1[tid + s];
    }
    __syncthreads();
  }
  if (tid == 0) {
    double sum = s0[0], sq = s1[0];
    double mean = sum / (double)E;
    double var = (sq - sum * sum / (double)E) / (double)(E - 1);
    if (var < 0.0) var = 0.0;
    double sd = sqrt(var);
    if (sd < 1e-5) sd = 1e-5;
    stats[0] = (float)mean;
    stats[1] = (float)(1.0 / sd);
  }
}

// ---------------------------------------------------------------------------
// Kernel 3: edge MLP 33->64->64->3 via WMMA f16.
// Each wave owns a private 16x64 activation tile in LDS and loops over MLP_T
// tiles, amortizing the one-time weight staging.  W1/W2 B-fragments are held
// in registers across the loop; W0 fragments stream from LDS.
// ---------------------------------------------------------------------------
__global__ void __launch_bounds__(128) llc_mlp_kernel(
    const float* __restrict__ scal, const float* __restrict__ ea,
    const float* __restrict__ stats, const float* __restrict__ W0,
    const float* __restrict__ b0, const float* __restrict__ W1,
    const float* __restrict__ b1, const float* __restrict__ W2,
    const float* __restrict__ b2, const int* __restrict__ src,
    const int* __restrict__ dst, float* __restrict__ logits, int E) {
  // B fragments pre-swizzled: tile = kt*4+nt -> [32 lanes][16 halves]
  __shared__ __align__(16) _Float16 w0s[8][512];  // 33x64 padded K->64
  __shared__ __align__(16) _Float16 w1s[8][512];  // 64x64
  __shared__ __align__(16) _Float16 w2s[2][512];  // 64x3 padded N->16
  __shared__ __align__(16) _Float16 xt[4][16 * 64];  // per-wave act tile

  const int tid = threadIdx.x;
  const int lane = tid & 31;
  const int wid = tid >> 5;

  // --- stage weights once per block (f32 -> f16, B-fragment lane order) ---
  for (int i = tid; i < 8 * 512; i += 128) {
    int tile = i >> 9, idx = i & 511;
    int l = idx >> 4, h = idx & 15;
    int kt = tile >> 2, nt = tile & 3;
    int k = kt * 32 + ((l & 16) ? 8 : 0) + (h < 8 ? h : h + 8);  // B: lanes>=16 hold K {8..15,24..31}
    int n = nt * 16 + (l & 15);
    w0s[tile][idx] = (_Float16)((k < 33) ? W0[k * 64 + n] : 0.f);
    w1s[tile][idx] = (_Float16)W1[k * 64 + n];
  }
  for (int i = tid; i < 2 * 512; i += 128) {
    int tile = i >> 9, idx = i & 511;
    int l = idx >> 4, h = idx & 15;
    int k = tile * 32 + ((l & 16) ? 8 : 0) + (h < 8 ? h : h + 8);
    int n = l & 15;
    w2s[tile][idx] = (_Float16)((n < 3) ? W2[k * 3 + n] : 0.f);
  }
  __syncthreads();  // the only block-wide barrier needed

  const float mean = stats[0], istd = stats[1];
  const int m = lane & 15;
  const int nn = lane & 15;
  const int mb = (lane & 16) ? 8 : 0;
  _Float16* xw = xt[wid];

  // --- loop-invariant registers: biases + W1/W2 B-fragments ---
  float b0v[4], b1v[4];
#pragma unroll
  for (int nt = 0; nt < 4; ++nt) {
    b0v[nt] = b0[nt * 16 + nn];
    b1v[nt] = b1[nt * 16 + nn];
  }
  float b2v = (nn < 3) ? b2[nn] : 0.f;
  v16h w1f[8];
#pragma unroll
  for (int t = 0; t < 8; ++t) w1f[t] = load_b_frag(w1s[t], lane);
  v16h w2f0 = load_b_frag(w2s[0], lane);
  v16h w2f1 = load_b_frag(w2s[1], lane);

  const int tile0 = (blockIdx.x * 4 + wid) * MLP_T;

  for (int t = 0; t < MLP_T; ++t) {
    const int ebase = (tile0 + t) * 16;
    const int e = min(ebase + m, E - 1);

    // --- stage x tile: [scalars_src(16) | scalars_dst(16) | ea | 0-pad] ---
    _Float16* xr = &xt[wid][m * 64];
    if (lane < 16) {
      int s = src[e];
#pragma unroll
      for (int k = 0; k < 16; ++k) xr[k] = (_Float16)scal[s * 16 + k];
      xr[32] = (_Float16)((ea[e] - mean) * istd);
#pragma unroll
      for (int k = 33; k < 64; ++k) xr[k] = (_Float16)0.f;
    } else {
      int d = dst[e];
#pragma unroll
      for (int k = 0; k < 16; ++k) xr[16 + k] = (_Float16)scal[d * 16 + k];
    }
    wave_lds_fence();

    // --- layer 0: (16x64) @ (64x64), W0 B-fragments from LDS ---
    v16h a0 = load_a_frag(xw, 0, lane);
    v16h a1 = load_a_frag(xw, 32, lane);
    v8f acc[4];
#pragma unroll
    for (int nt = 0; nt < 4; ++nt) {
      v8f c = {};
      c = wmma_f16(a0, load_b_frag(w0s[nt], lane), c);
      c = wmma_f16(a1, load_b_frag(w0s[4 + nt], lane), c);
      acc[nt] = c;
    }
    wave_lds_fence();
#pragma unroll
    for (int nt = 0; nt < 4; ++nt) {
#pragma unroll
      for (int r = 0; r < 8; ++r)
        xt[wid][(mb + r) * 64 + nt * 16 + nn] =
            (_Float16)fmaxf(acc[nt][r] + b0v[nt], 0.f);
    }
    wave_lds_fence();

    // --- layer 1: (16x64) @ (64x64), W1 B-fragments in registers ---
    a0 = load_a_frag(xw, 0, lane);
    a1 = load_a_frag(xw, 32, lane);
#pragma unroll
    for (int nt = 0; nt < 4; ++nt) {
      v8f c = {};
      c = wmma_f16(a0, w1f[nt], c);
      c = wmma_f16(a1, w1f[4 + nt], c);
      acc[nt] = c;
    }
    wave_lds_fence();
#pragma unroll
    for (int nt = 0; nt < 4; ++nt) {
#pragma unroll
      for (int r = 0; r < 8; ++r)
        xt[wid][(mb + r) * 64 + nt * 16 + nn] =
            (_Float16)fmaxf(acc[nt][r] + b1v[nt], 0.f);
    }
    wave_lds_fence();

    // --- layer 2: (16x64) @ (64x16 padded, 3 valid cols) ---
    a0 = load_a_frag(xw, 0, lane);
    a1 = load_a_frag(xw, 32, lane);
    v8f c2 = {};
    c2 = wmma_f16(a0, w2f0, c2);
    c2 = wmma_f16(a1, w2f1, c2);
    if (nn < 3) {
#pragma unroll
      for (int r = 0; r < 8; ++r) {
        int edge = ebase + mb + r;
        if (edge < E) logits[(size_t)edge * 3 + nn] = c2[r] + b2v;
      }
    }
    wave_lds_fence();  // next iteration overwrites xt
  }
}

// ---------------------------------------------------------------------------
// Kernel 4: per-node softmax over 31 contiguous edges + weighted 4-vec sums
// ---------------------------------------------------------------------------
__global__ void __launch_bounds__(256) llc_softmax_kernel(
    const float* __restrict__ p4, const float* __restrict__ logits,
    const int* __restrict__ src, const int* __restrict__ dst,
    float* __restrict__ vecs, int N, int epn) {
  int lane = threadIdx.x & 31, wid = threadIdx.x >> 5;
  int node = blockIdx.x * 8 + wid;
  if (node >= N) return;
  bool valid = lane < epn;
  int e = node * epn + (valid ? lane : epn - 1);
  float l0 = logits[(size_t)e * 3 + 0];
  float l1 = logits[(size_t)e * 3 + 1];
  float l2 = logits[(size_t)e * 3 + 2];
  const float NEG = -3.0e38f;
  float m0 = wave_max(valid ? l0 : NEG);
  float m1 = wave_max(valid ? l1 : NEG);
  float m2 = wave_max(valid ? l2 : NEG);
  float x0 = valid ? expf(l0 - m0) : 0.f;
  float x1 = valid ? expf(l1 - m1) : 0.f;
  float x2 = valid ? expf(l2 - m2) : 0.f;
  float s0 = wave_sum(x0), s1 = wave_sum(x1), s2 = wave_sum(x2);
  float wv[3];
  wv[0] = x0 / fmaxf(s0, 1e-16f);
  wv[1] = x1 / fmaxf(s1, 1e-16f);
  wv[2] = x2 / fmaxf(s2, 1e-16f);
  int si = src[e], di = dst[e];
  float fr[4];
#pragma unroll
  for (int d = 0; d < 4; ++d) fr[d] = p4[si * 4 + d] + p4[di * 4 + d];
  float lf = fr[0] * fr[0] - fr[1] * fr[1] - fr[2] * fr[2] - fr[3] * fr[3];
  float inv = 1.f / fmaxf(sqrtf(fabsf(lf)), 1e-6f);
#pragma unroll
  for (int d = 0; d < 4; ++d) fr[d] *= inv;
#pragma unroll
  for (int c = 0; c < 3; ++c) {
#pragma unroll
    for (int d = 0; d < 4; ++d) {
      float t = valid ? wv[c] * fr[d] : 0.f;
      t = wave_sum(t);
      if (lane == 0) vecs[(size_t)node * 12 + c * 4 + d] = t;
    }
  }
}

// ---------------------------------------------------------------------------
// Kernel 5: per-node frame construction (clamp-boost + polar decomposition)
// ---------------------------------------------------------------------------
__global__ void __launch_bounds__(256) llc_frame_kernel(
    const float* __restrict__ vecs, float* __restrict__ out, int N) {
  int n = blockIdx.x * 256 + threadIdx.x;
  if (n >= N) return;
  float v[3][4];
#pragma unroll
  for (int c = 0; c < 3; ++c)
#pragma unroll
    for (int d = 0; d < 4; ++d) v[c][d] = vecs[(size_t)n * 12 + c * 4 + d];
  float ln = 0.f;
#pragma unroll
  for (int c = 0; c < 3; ++c)
    ln += v[c][0] * v[c][0] - v[c][1] * v[c][1] - v[c][2] * v[c][2] -
          v[c][3] * v[c][3];
  float s = 1.f / fmaxf(sqrtf(fabsf(ln)), 1e-5f);
#pragma unroll
  for (int c = 0; c < 3; ++c)
#pragma unroll
    for (int d = 0; d < 4; ++d) v[c][d] *= s;

  // _clamp_boost(v[0])
  float lv = v[0][0] * v[0][0] - v[0][1] * v[0][1] - v[0][2] * v[0][2] -
             v[0][3] * v[0][3];
  float mass = sqrtf(fmaxf(lv, 0.f));
  float t0 = v[0][0];
  float it0 = 1.f / fmaxf(t0, 1e-10f);
  float be[3] = {v[0][1] * it0, v[0][2] * it0, v[0][3] * it0};
  float gamma = t0 / fmaxf(mass, 1e-10f);
  float g = fminf(fmaxf(gamma, 1.f), GAMMA_MAX);
  float gc = fmaxf(g, 1e-10f);
  float num = sqrtf(fmaxf(1.f - 1.f / (gc * gc), 1e-10f));
  float den =
      sqrtf(fmaxf(be[0] * be[0] + be[1] * be[1] + be[2] * be[2], 1e-10f));
  float bsc = num / den;
  float fmv[4];
  fmv[0] = mass * g;
  fmv[1] = mass * g * be[0] * bsc;
  fmv[2] = mass * g * be[1] * bsc;
  fmv[3] = mass * g * be[2] * bsc;

  // _reg_lightlike (deterministic surrogate; mask essentially never fires)
  float lfm = fmv[0] * fmv[0] - fmv[1] * fmv[1] - fmv[2] * fmv[2] -
              fmv[3] * fmv[3];
  if (fabsf(lfm) < EPS32) {
    const float r1 = 0.79788f, r2 = 0.79788f, r3 = 0.79788f;
    float r0 = sqrtf(2.f * (r1 * r1 + r2 * r2 + r3 * r3));
    fmv[0] += EPS32 * r0;
    fmv[1] += EPS32 * r1;
    fmv[2] += EPS32 * r2;
    fmv[3] += EPS32 * r3;
  }

  // _restframe_boost
  float tt = fmv[0];
  float itt = 1.f / fmaxf(tt, 1e-10f);
  float bv[3] = {fmv[1] * itt, fmv[2] * itt, fmv[3] * itt};
  float bb2 = bv[0] * bv[0] + bv[1] * bv[1] + bv[2] * bv[2];
  float gam = 1.f / sqrtf(fmaxf(1.f - bb2, 1e-10f));
  float bo[3] = {-gam * bv[0], -gam * bv[1], -gam * bv[2]};
  float sc = (gam - 1.f) / fmaxf(bb2, 1e-10f);
  float B[4][4];
  B[0][0] = gam;
  B[0][1] = bo[0];
  B[0][2] = bo[1];
  B[0][3] = bo[2];
#pragma unroll
  for (int i = 0; i < 3; ++i) {
    B[i + 1][0] = bo[i];
#pragma unroll
    for (int j = 0; j < 3; ++j)
      B[i + 1][j + 1] = ((i == j) ? 1.f : 0.f) + sc * bv[i] * bv[j];
  }

  // rr[r][k] = sum_d refs[r][d] * B[k][d];   v0,v1 = spatial parts
  float v0[3], v1[3];
#pragma unroll
  for (int k = 0; k < 3; ++k) {
    v0[k] = B[k + 1][0] * v[1][0] + B[k + 1][1] * v[1][1] +
            B[k + 1][2] * v[1][2] + B[k + 1][3] * v[1][3];
    v1[k] = B[k + 1][0] * v[2][0] + B[k + 1][1] * v[2][1] +
            B[k + 1][2] * v[2][2] + B[k + 1][3] * v[2][3];
  }
  float cr0 = v0[1] * v1[2] - v0[2] * v1[1];
  float cr1 = v0[2] * v1[0] - v0[0] * v1[2];
  float cr2 = v0[0] * v1[1] - v0[1] * v1[0];
  if (cr0 * cr0 + cr1 * cr1 + cr2 * cr2 < EPS32) {
    v0[0] += EPS32 * 0.31f; v0[1] -= EPS32 * 0.77f; v0[2] += EPS32 * 1.13f;
    v1[0] -= EPS32 * 0.59f; v1[1] += EPS32 * 0.41f; v1[2] -= EPS32 * 0.97f;
  }
  float n0 = sqrtf(v0[0] * v0[0] + v0[1] * v0[1] + v0[2] * v0[2]);
  float i0 = 1.f / fmaxf(n0, EPS32);
  float e0[3] = {v0[0] * i0, v0[1] * i0, v0[2] * i0};
  float n1 = sqrtf(v1[0] * v1[0] + v1[1] * v1[1] + v1[2] * v1[2]);
  float i1 = 1.f / fmaxf(n1, EPS32);
  float v1n[3] = {v1[0] * i1, v1[1] * i1, v1[2] * i1};
  float dp = v1n[0] * e0[0] + v1n[1] * e0[1] + v1n[2] * e0[2];
  float u1[3] = {v1n[0] - dp * e0[0], v1n[1] - dp * e0[1], v1n[2] - dp * e0[2]};
  float nu = sqrtf(u1[0] * u1[0] + u1[1] * u1[1] + u1[2] * u1[2]);
  float iu = 1.f / fmaxf(nu, EPS32);
  float e1[3] = {u1[0] * iu, u1[1] * iu, u1[2] * iu};
  float e2[3] = {e0[1] * e1[2] - e0[2] * e1[1], e0[2] * e1[0] - e0[0] * e1[2],
                 e0[0] * e1[1] - e0[1] * e1[0]};

  float* o = out + (size_t)n * 16;
#pragma unroll
  for (int k = 0; k < 4; ++k) o[k] = B[0][k];
  const float* Ev[3] = {e0, e1, e2};
#pragma unroll
  for (int i = 0; i < 3; ++i)
#pragma unroll
    for (int k = 0; k < 4; ++k)
      o[(i + 1) * 4 + k] =
          Ev[i][0] * B[1][k] + Ev[i][1] * B[2][k] + Ev[i][2] * B[3][k];
}

// ---------------------------------------------------------------------------
extern "C" void kernel_launch(void* const* d_in, const int* in_sizes, int n_in,
                              void* d_out, int out_size, void* d_ws,
                              size_t ws_size, hipStream_t stream) {
  const float* p4 = (const float*)d_in[0];
  const float* scal = (const float*)d_in[1];
  const int* src = (const int*)d_in[2];
  const int* dst = (const int*)d_in[3];
  const float* W0 = (const float*)d_in[4];
  const float* b0 = (const float*)d_in[5];
  const float* W1 = (const float*)d_in[6];
  const float* b1 = (const float*)d_in[7];
  const float* W2 = (const float*)d_in[8];
  const float* b2 = (const float*)d_in[9];

  const int E = in_sizes[2];      // number of edges
  const int N = in_sizes[0] / 4;  // number of particles
  const int epn = E / N;          // edges per source node (31)

  // workspace layout
  float* ws_f = (float*)d_ws;
  float* ea = ws_f;                      // E floats
  float* logits = ea + (size_t)E;        // 3E floats
  float* vecs = logits + (size_t)3 * E;  // 12N floats
  float* stats = vecs + (size_t)12 * N;  // 2 floats
  size_t paddr = ((size_t)(stats + 2) + 7) & ~(size_t)7;
  double* partials = (double*)paddr;     // 2 * nb1 doubles

  const int nb1 = (E + 255) / 256;
  llc_ea_kernel<<<nb1, 256, 0, stream>>>(p4, src, dst, ea, partials, E);
  llc_stats_kernel<<<1, 256, 0, stream>>>(partials, nb1, (long long)E, stats);

  const int tiles = (E + 15) / 16;
  const int waves = (tiles + MLP_T - 1) / MLP_T;
  const int nb3 = (waves + 3) / 4;  // 4 waves per 128-thread block
  llc_mlp_kernel<<<nb3, 128, 0, stream>>>(scal, ea, stats, W0, b0, W1, b1, W2,
                                          b2, src, dst, logits, E);

  const int nb4 = (N + 7) / 8;  // 8 waves (nodes) per 256-thread block
  llc_softmax_kernel<<<nb4, 256, 0, stream>>>(p4, logits, src, dst, vecs, N,
                                              epn);

  const int nb5 = (N + 255) / 256;
  llc_frame_kernel<<<nb5, 256, 0, stream>>>(vecs, (float*)d_out, N);
}
